// NGPColorField_37074157699563
// MI455X (gfx1250) — matrix operand order repaired and checked by reading, not compiled, not run on hardware
//
#include <hip/hip_runtime.h>
#include <cstdint>

typedef _Float16 v16h __attribute__((ext_vector_type(16)));
typedef float    v8f  __attribute__((ext_vector_type(8)));

#define WAVES_PER_BLOCK 8
#define ITERS 8
#define PTS_PER_BLOCK (WAVES_PER_BLOCK * ITERS * 16)   // 1024 points per block

__global__ __launch_bounds__(256)
void ngp_field_kernel(const float* __restrict__ positions,
                      const float* __restrict__ table,
                      const float* __restrict__ W1d, const float* __restrict__ b1d,
                      const float* __restrict__ W2d, const float* __restrict__ b2d,
                      const float* __restrict__ W1c, const float* __restrict__ b1c,
                      const float* __restrict__ W2c, const float* __restrict__ b2c,
                      float* __restrict__ out, int N)
{
    // ---- staged weights (f32) ----
    __shared__ float sW1d[2048];   // [32][64]
    __shared__ float sW1c[2048];
    __shared__ float sW2d[64];
    __shared__ float sW2c[192];    // [64][3]
    __shared__ float sB1d[64];
    __shared__ float sB1c[64];
    __shared__ float sB2[4];       // [b2d, b2c0, b2c1, b2c2]

    const int tid = threadIdx.x;
    for (int i = tid; i < 2048; i += 256) { sW1d[i] = W1d[i]; sW1c[i] = W1c[i]; }
    for (int i = tid; i < 192;  i += 256) { sW2c[i] = W2c[i]; }
    if (tid < 64) { sW2d[tid] = W2d[tid]; sB1d[tid] = b1d[tid]; sB1c[tid] = b1c[tid]; }
    if (tid < 3)  { sB2[1 + tid] = b2c[tid]; }
    if (tid == 3) { sB2[0] = b2d[0]; }
    __syncthreads();

    const int lane = tid & 31;
    const int wv   = tid >> 5;
    const int n    = lane & 15;     // point row (encode) / matrix column (WMMA B,D)
    const int hi   = lane >> 4;     // which half-wave

    // ---- build f16 B-matrices in registers (16-bit layout: half h holds
    //      K=8h..8h+7 then K=16+8h..16+8h+7; VGPR v packs K=k0, k0+1 of column n) ----
    v16h Bd[4], Bc[4];
    #pragma unroll
    for (int j = 0; j < 4; ++j) {
        #pragma unroll
        for (int v = 0; v < 8; ++v) {
            const int k0 = hi * 8 + 2 * v + ((v >= 4) ? 8 : 0);
            const int col = j * 16 + n;
            Bd[j][2*v]   = (_Float16)sW1d[(k0    ) * 64 + col];
            Bd[j][2*v+1] = (_Float16)sW1d[(k0 + 1) * 64 + col];
            Bc[j][2*v]   = (_Float16)sW1c[(k0    ) * 64 + col];
            Bc[j][2*v+1] = (_Float16)sW1c[(k0 + 1) * 64 + col];
        }
    }
    // per-lane layer-2 weights / layer-1 biases (lane owns hidden unit 16j+n)
    float b1dl[4], b1cl[4], w2dl[4], w2cl[4][3];
    #pragma unroll
    for (int j = 0; j < 4; ++j) {
        const int u = j * 16 + n;
        b1dl[j] = sB1d[u];  b1cl[j] = sB1c[u];  w2dl[j] = sW2d[u];
        w2cl[j][0] = sW2c[u*3 + 0];
        w2cl[j][1] = sW2c[u*3 + 1];
        w2cl[j][2] = sW2c[u*3 + 2];
    }
    const float b2dr = sB2[0], b2c0 = sB2[1], b2c1 = sB2[2], b2c2 = sB2[3];

    // Level resolutions (floor(16 * 256^(l/15))). Lane handles levels
    // {4hi..4hi+3, 8+4hi..8+4hi+3} so its packed accumulators are exactly its
    // WMMA A-matrix registers (16-bit A layout).
    const float RL[8] = {16.f, 23.f, 33.f, 48.f, 307.f, 445.f, 645.f, 933.f};
    const float RH[8] = {70.f, 101.f, 147.f, 212.f, 1351.f, 1955.f, 2830.f, 4096.f};

    const uint32_t P1 = 2654435761u, P2 = 805459861u, MK = (1u << 19) - 1u;
    const float2* __restrict__ tab2 = reinterpret_cast<const float2*>(table);

    const int waveBase = blockIdx.x * PTS_PER_BLOCK + wv * (ITERS * 16);
    const v8f zc = {0.f, 0.f, 0.f, 0.f, 0.f, 0.f, 0.f, 0.f};

    // fold-select bits for the tree reduction
    const bool sel8 = (lane & 8) != 0;
    const bool sel4 = (lane & 4) != 0;
    const bool sel2 = (lane & 2) != 0;
    // point this lane finalizes after the tree: m = 8*hi + ((lane>>1)&7)
    const int mfin   = 8 * hi + ((lane >> 1) & 7);
    const int srcln  = 16 * hi + mfin;   // lane that encoded point mfin (same half)

    // tree-reduce 8 per-lane partials across each 16-lane half; returns the
    // completed sum for point mfin (duplicated on lane pairs {2k,2k+1})
    auto tree16 = [&](float p[8]) -> float {
        #pragma unroll
        for (int i = 0; i < 8; ++i) p[i] += __shfl_xor(p[i], 8, 32);
        float q4[4];
        #pragma unroll
        for (int i = 0; i < 4; ++i) q4[i] = sel8 ? p[i + 4] : p[i];
        #pragma unroll
        for (int i = 0; i < 4; ++i) q4[i] += __shfl_xor(q4[i], 4, 32);
        float q2[2];
        q2[0] = sel4 ? q4[2] : q4[0];
        q2[1] = sel4 ? q4[3] : q4[1];
        q2[0] += __shfl_xor(q2[0], 2, 32);
        q2[1] += __shfl_xor(q2[1], 2, 32);
        float q1 = sel2 ? q2[1] : q2[0];
        q1 += __shfl_xor(q1, 1, 32);
        return q1;
    };

    for (int it = 0; it < ITERS; ++it) {
        const int iterBase = waveBase + it * 16;
        int pt = iterBase + n; if (pt >= N) pt = N - 1;   // clamp: keep flow uniform

        const float* pp = positions + 3 * (size_t)pt;
        const float x = pp[0], y = pp[1], z = pp[2];
        const float xn0 = (x + 1.f) * 0.5f;
        const float xn1 = (y + 1.f) * 0.5f;
        const float xn2 = (z + 1.f) * 0.5f;
        const float bias = 10.f - 20.f * sqrtf(x*x + y*y + z*z);
        const float selv = (xn0 > 0.f && xn0 < 1.f &&
                            xn1 > 0.f && xn1 < 1.f &&
                            xn2 > 0.f && xn2 < 1.f) ? 1.f : 0.f;

        // ---- hash-grid encode: 8 levels per lane, 8 corners each ----
        v16h A;
        #pragma unroll
        for (int s = 0; s < 8; ++s) {
            const float R = hi ? RH[s] : RL[s];
            const int lvl = ((s < 4) ? s : (s + 4)) + 4 * hi;
            const float2* __restrict__ tl = tab2 + ((size_t)lvl << 19);

            const float px = xn0 * R, py = xn1 * R, pz = xn2 * R;
            const float fx = floorf(px), fy = floorf(py), fz = floorf(pz);
            const float wx = px - fx, wy = py - fy, wz = pz - fz;
            const uint32_t ix = (uint32_t)fx, iy = (uint32_t)fy, iz = (uint32_t)fz;
            const uint32_t hx0 = ix,        hx1 = ix + 1u;
            const uint32_t hy0 = iy * P1,   hy1 = hy0 + P1;
            const uint32_t hz0 = iz * P2,   hz1 = hz0 + P2;
            const uint32_t g00 = hx0 ^ hy0, g10 = hx1 ^ hy0;
            const uint32_t g01 = hx0 ^ hy1, g11 = hx1 ^ hy1;
            const float wx0 = 1.f - wx, wy0 = 1.f - wy, wz0 = 1.f - wz;
            const float w00 = wx0 * wy0, w10 = wx * wy0;
            const float w01 = wx0 * wy,  w11 = wx * wy;

            float a0 = 0.f, a1 = 0.f;
            float2 t; float wt;
            t = tl[(g00 ^ hz0) & MK]; wt = w00 * wz0; a0 = fmaf(t.x, wt, a0); a1 = fmaf(t.y, wt, a1);
            t = tl[(g10 ^ hz0) & MK]; wt = w10 * wz0; a0 = fmaf(t.x, wt, a0); a1 = fmaf(t.y, wt, a1);
            t = tl[(g01 ^ hz0) & MK]; wt = w01 * wz0; a0 = fmaf(t.x, wt, a0); a1 = fmaf(t.y, wt, a1);
            t = tl[(g11 ^ hz0) & MK]; wt = w11 * wz0; a0 = fmaf(t.x, wt, a0); a1 = fmaf(t.y, wt, a1);
            t = tl[(g00 ^ hz1) & MK]; wt = w00 * wz;  a0 = fmaf(t.x, wt, a0); a1 = fmaf(t.y, wt, a1);
            t = tl[(g10 ^ hz1) & MK]; wt = w10 * wz;  a0 = fmaf(t.x, wt, a0); a1 = fmaf(t.y, wt, a1);
            t = tl[(g01 ^ hz1) & MK]; wt = w01 * wz;  a0 = fmaf(t.x, wt, a0); a1 = fmaf(t.y, wt, a1);
            t = tl[(g11 ^ hz1) & MK]; wt = w11 * wz;  a0 = fmaf(t.x, wt, a0); a1 = fmaf(t.y, wt, a1);

            A[2*s]     = (_Float16)a0;
            A[2*s + 1] = (_Float16)a1;
        }

        // ---- layer 1 via WMMA, layer 2 via lane-FMA partials ----
        float pd[8] = {0,0,0,0,0,0,0,0};
        float pr[8] = {0,0,0,0,0,0,0,0};
        float pg[8] = {0,0,0,0,0,0,0,0};
        float pb[8] = {0,0,0,0,0,0,0,0};

        #pragma unroll
        for (int j = 0; j < 4; ++j) {
            v8f D = __builtin_amdgcn_wmma_f32_16x16x32_f16(
                        false, A, false, Bd[j], (short)0, zc, false, false);
            #pragma unroll
            for (int v = 0; v < 8; ++v) {
                const float h = fmaxf(D[v] + b1dl[j], 0.f);
                pd[v] = fmaf(h, w2dl[j], pd[v]);
            }
        }
        #pragma unroll
        for (int j = 0; j < 4; ++j) {
            v8f D = __builtin_amdgcn_wmma_f32_16x16x32_f16(
                        false, A, false, Bc[j], (short)0, zc, false, false);
            #pragma unroll
            for (int v = 0; v < 8; ++v) {
                const float h = fmaxf(D[v] + b1cl[j], 0.f);
                pr[v] = fmaf(h, w2cl[j][0], pr[v]);
                pg[v] = fmaf(h, w2cl[j][1], pg[v]);
                pb[v] = fmaf(h, w2cl[j][2], pb[v]);
            }
        }

        // ---- folding-tree reduction over the 16 hidden columns in each half;
        //      lane ends holding the full sums for point mfin ----
        const float fd = tree16(pd);
        const float fr = tree16(pr);
        const float fg = tree16(pg);
        const float fb = tree16(pb);

        // fetch blob-bias / selector of point mfin from the lane that encoded it
        const float biasM = __shfl(bias, srcln, 32);
        const float selvM = __shfl(selv, srcln, 32);

        // even lanes finalize one point each (all 16 points covered, no LDS bounce)
        if ((lane & 1) == 0) {
            const int p2 = iterBase + mfin;
            if (p2 < N) {
                const float dpre = fd + b2dr + biasM;
                float dens = (dpre > 20.f) ? dpre : log1pf(__expf(dpre));
                dens *= selvM;
                const float r = 1.f / (1.f + __expf(-(fr + b2c0)));
                const float g = 1.f / (1.f + __expf(-(fg + b2c1)));
                const float b = 1.f / (1.f + __expf(-(fb + b2c2)));
                float* o = out + 3 * (size_t)p2;
                o[0] = r; o[1] = g; o[2] = b;
                out[3 * (size_t)N + p2] = dens;
            }
        }
    }
    (void)wv;
}

extern "C" void kernel_launch(void* const* d_in, const int* in_sizes, int n_in,
                              void* d_out, int out_size, void* d_ws, size_t ws_size,
                              hipStream_t stream) {
    (void)n_in; (void)out_size; (void)d_ws; (void)ws_size;
    const int N = in_sizes[0] / 3;
    const int grid = (N + PTS_PER_BLOCK - 1) / PTS_PER_BLOCK;
    ngp_field_kernel<<<grid, 256, 0, stream>>>(
        (const float*)d_in[0],  // positions
        (const float*)d_in[1],  // table
        (const float*)d_in[2], (const float*)d_in[3],   // W1d, b1d
        (const float*)d_in[4], (const float*)d_in[5],   // W2d, b2d
        (const float*)d_in[6], (const float*)d_in[7],   // W1c, b1c
        (const float*)d_in[8], (const float*)d_in[9],   // W2c, b2c
        (float*)d_out, N);
}
